// MonolithicSSM_62302795596483
// MI455X (gfx1250) — matrix operand order repaired and checked by reading, not compiled, not run on hardware
//
#include <hip/hip_runtime.h>
#include <hip/hip_bf16.h>
#include <math.h>

typedef __attribute__((ext_vector_type(16))) _Float16 v16h;
typedef __attribute__((ext_vector_type(8)))  float    v8f;

#define D_IN     64
#define D_MODEL  256
#define N_STATE  32
#define N_LAYERS 4
#define N_CLS    10
#define BB       8
#define SS       8192
#define ROWS     (BB*SS)      /* 65536 */
#define CHUNK    512
#define NCH      (SS/CHUNK)   /* 16 */
#define EPSV     1e-8f

// ---------------- f32 -> f16 convert ----------------
__global__ void cvt_f16_kernel(const float* __restrict__ src, _Float16* __restrict__ dst, int n) {
  int i = blockIdx.x * blockDim.x + threadIdx.x;
  if (i < n) dst[i] = (_Float16)src[i];
}

// ---- CDNA5 async global->LDS copy (ASYNCcnt path, no VGPR round trip) ----
__device__ __forceinline__ void async_copy_b128(const _Float16* __restrict__ g,
                                                _Float16* l) {
  // LDS operand: low 32 bits of the generic pointer == wave-relative LDS offset.
  unsigned lds = (unsigned)(unsigned long long)l;
  unsigned long long ga = (unsigned long long)g;
  asm volatile("global_load_async_to_lds_b128 %0, %1, off"
               :: "v"(lds), "v"(ga) : "memory");
}
__device__ __forceinline__ void async_wait0() {
  asm volatile("s_wait_asynccnt 0x0" ::: "memory");
}

// ---------------- WMMA GEMM ----------------
// C[M,N](f32) = A[M,K](f16, lda) * BT[N,K](f16, ldb)^T + bias[N] (+ residual[M,N])
// Block: 128 threads (4 waves). Block tile: 128(M) x 32(N). Wave tile: 32x32
// (4 independent WMMA accumulator chains). Tiles staged with
// global_load_async_to_lds_b128; epilogue staged through LDS, float4 stores.
// Requires: M % 128 == 0, N % 32 == 0, K % 32 == 0.
__global__ __launch_bounds__(128) void wmma_gemm_kernel(
    const _Float16* __restrict__ A, int lda,
    const _Float16* __restrict__ BT, int ldb,
    const float* __restrict__ bias,
    const float* __restrict__ residual, int ldr,
    float* __restrict__ C, int ldc,
    int K)
{
  __shared__ __align__(16) _Float16 sA[128 * 32];  // 8 KB
  __shared__ __align__(16) _Float16 sB[32 * 32];   // 2 KB
  __shared__ __align__(16) float    sO[128 * 32];  // 16 KB epilogue staging

  const int tid  = threadIdx.x;
  const int wave = tid >> 5;
  const int lane = tid & 31;
  const int l    = lane & 15;
  const int hb   = lane >> 4;          // lane half: 0 or 1
  const long m0  = (long)blockIdx.x * 128;
  const long n0  = (long)blockIdx.y * 32;

  // acc[ms][ns]; column of acc[*][0] is n0+l, of acc[*][1] is n0+16+l
  v8f acc00, acc01, acc10, acc11;
  const float b0 = bias[n0 + l];
  const float b1 = bias[n0 + 16 + l];
#pragma unroll
  for (int r = 0; r < 8; ++r) {
    acc00[r] = b0; acc10[r] = b0;
    acc01[r] = b1; acc11[r] = b1;
  }

  for (int k0 = 0; k0 < K; k0 += 32) {
    __syncthreads();   // previous iteration's LDS reads complete
    // Stage A tile 128x32 halves: 512 x b128, 4 per thread (async to LDS)
    {
      int idx = tid;
#pragma unroll
      for (int it = 0; it < 4; ++it, idx += 128) {
        int row = idx >> 2, c4 = idx & 3;
        async_copy_b128(A + (m0 + row) * (long)lda + k0 + c4 * 8,
                        &sA[row * 32 + c4 * 8]);
      }
      // Stage B tile 32(N) x 32(K) halves: 128 x b128, 1 per thread
      int row = tid >> 2, c4 = tid & 3;
      async_copy_b128(BT + (n0 + row) * (long)ldb + k0 + c4 * 8,
                      &sB[row * 32 + c4 * 8]);
    }
    async_wait0();     // drain this wave's ASYNCcnt
    __syncthreads();   // all waves' async LDS writes now visible

    if (k0 + 32 < K)   // WGP-scope prefetch of next A tile (global_prefetch)
      __builtin_prefetch(A + (m0 + tid) * (long)lda + k0 + 32, 0, 3);

    // Build fragments per CDNA5 ISA VGPR layouts
    union { uint4 u[2]; v16h h; } fa0, fa1, fb0, fb1;
    const uint4* sA4 = (const uint4*)sA;
    const uint4* sB4 = (const uint4*)sB;
    // A 16x32: lane row = l; K = hb*8+[0..7] then 16+hb*8+[0..7]
    const int mA0 = wave * 32 + l;
    const int mA1 = wave * 32 + 16 + l;
    fa0.u[0] = sA4[mA0 * 4 + hb];
    fa0.u[1] = sA4[mA0 * 4 + 2 + hb];
    fa1.u[0] = sA4[mA1 * 4 + hb];
    fa1.u[1] = sA4[mA1 * 4 + 2 + hb];
    // B 32x16: lane col = l; K = hb*8+[0..7] then 16+hb*8+[0..7]
    fb0.u[0] = sB4[l * 4 + hb];
    fb0.u[1] = sB4[l * 4 + 2 + hb];
    fb1.u[0] = sB4[(16 + l) * 4 + hb];
    fb1.u[1] = sB4[(16 + l) * 4 + 2 + hb];

    acc00 = __builtin_amdgcn_wmma_f32_16x16x32_f16(false, fa0.h, false, fb0.h,
                                                   (short)0, acc00, false, false);
    acc01 = __builtin_amdgcn_wmma_f32_16x16x32_f16(false, fa0.h, false, fb1.h,
                                                   (short)0, acc01, false, false);
    acc10 = __builtin_amdgcn_wmma_f32_16x16x32_f16(false, fa1.h, false, fb0.h,
                                                   (short)0, acc10, false, false);
    acc11 = __builtin_amdgcn_wmma_f32_16x16x32_f16(false, fa1.h, false, fb1.h,
                                                   (short)0, acc11, false, false);
  }

  // ---- Epilogue: D layout lane col = l (+16), VGPR r -> M = r + 8*hb ----
  __syncthreads();
  {
    const int rbase = wave * 32 + hb * 8;
#pragma unroll
    for (int r = 0; r < 8; ++r) {
      sO[(rbase + r)      * 32 + l]      = acc00[r];
      sO[(rbase + r)      * 32 + 16 + l] = acc01[r];
      sO[(rbase + 16 + r) * 32 + l]      = acc10[r];
      sO[(rbase + 16 + r) * 32 + 16 + l] = acc11[r];
    }
  }
  __syncthreads();
  // 128 rows x 32 floats = 1024 float4 segments; 8 per thread, coalesced.
  {
    int u = tid;
#pragma unroll
    for (int it = 0; it < 8; ++it, u += 128) {
      int row = u >> 3, c4 = u & 7;
      float4 v = *(const float4*)&sO[row * 32 + c4 * 4];
      long gcol = n0 + c4 * 4;
      if (residual) {
        float4 rv = *(const float4*)(residual + (m0 + row) * (long)ldr + gcol);
        v.x += rv.x; v.y += rv.y; v.z += rv.z; v.w += rv.w;
      }
      *(float4*)(C + (m0 + row) * (long)ldc + gcol) = v;
    }
  }
}

// ---------------- block reduction helper (wave32 shuffle + tiny LDS) ------
__device__ inline float block256_reduce(float x) {
#pragma unroll
  for (int o = 16; o > 0; o >>= 1) x += __shfl_xor(x, o, 32);
  __shared__ float w[8];
  int wid = threadIdx.x >> 5, ln = threadIdx.x & 31;
  if (ln == 0) w[wid] = x;
  __syncthreads();
  float t = 0.f;
#pragma unroll
  for (int i = 0; i < 8; ++i) t += w[i];
  return t;
}

// ---------------- rmsnorm -> f16 ----------------
__global__ __launch_bounds__(256) void rmsnorm_f16_kernel(
    const float* __restrict__ h, const float* __restrict__ scale,
    _Float16* __restrict__ out)
{
  long row = blockIdx.x;
  int d = threadIdx.x;
  float v = h[row * D_MODEL + d];
  float tot = block256_reduce(v * v);
  float norm = sqrtf(tot * (1.0f / D_MODEL));
  out[row * D_MODEL + d] = (_Float16)(scale[d] * v / (norm + EPSV));
}

// ---------------- per-row inverse norm (final rmsnorm) ----------------
__global__ __launch_bounds__(256) void rownorm_kernel(
    const float* __restrict__ h, float* __restrict__ rinv)
{
  long row = blockIdx.x;
  int d = threadIdx.x;
  float v = h[row * D_MODEL + d];
  float tot = block256_reduce(v * v);
  if (d == 0) rinv[row] = 1.0f / (sqrtf(tot * (1.0f / D_MODEL)) + EPSV);
}

// ---------------- mean over S of scale[d]*h*rinv ----------------
__global__ __launch_bounds__(256) void mean_kernel(
    const float* __restrict__ h, const float* __restrict__ rinv,
    const float* __restrict__ fscale, float* __restrict__ hmean)
{
  int b = blockIdx.x, d = threadIdx.x;
  const float* hb = h + (long)b * SS * D_MODEL;
  const float* rb = rinv + (long)b * SS;
  float acc = 0.f;
  for (int s = 0; s < SS; ++s) acc += hb[(long)s * D_MODEL + d] * rb[s];
  hmean[b * D_MODEL + d] = fscale[d] * acc * (1.0f / SS);
}

// ---------------- chunked parallel scan ----------------
// Phase 1: local scan per chunk (zero init), in-place over bx; emits chunk carries.
__global__ __launch_bounds__(32) void scan_local_kernel(
    float* __restrict__ bx, const float* __restrict__ logA, float* __restrict__ carry)
{
  int b = blockIdx.x, c = blockIdx.y, n = threadIdx.x;
  float A = 1.0f / (1.0f + expf(-logA[n]));
  long base = ((long)b * SS + (long)c * CHUNK) * N_STATE + n;
  float hst = 0.f;
  for (int t = 0; t < CHUNK; ++t) {
    hst = A * hst + bx[base + (long)t * N_STATE];
    bx[base + (long)t * N_STATE] = hst;
  }
  carry[(b * NCH + c) * N_STATE + n] = hst;
}

// Phase 2: exclusive scan of chunk carries with factor A^CHUNK.
__global__ __launch_bounds__(32) void scan_combine_kernel(
    const float* __restrict__ carry, const float* __restrict__ logA,
    float* __restrict__ prefix)
{
  int b = blockIdx.x, n = threadIdx.x;
  float A  = 1.0f / (1.0f + expf(-logA[n]));
  float AC = powf(A, (float)CHUNK);
  float P = 0.f;
  for (int c = 0; c < NCH; ++c) {
    prefix[(b * NCH + c) * N_STATE + n] = P;
    P = AC * P + carry[(b * NCH + c) * N_STATE + n];
  }
}

// Phase 3: hs_t = local_t + A^(t+1) * prefix ; write f16 for next GEMM.
__global__ __launch_bounds__(32) void scan_apply_kernel(
    const float* __restrict__ bx, const float* __restrict__ logA,
    const float* __restrict__ prefix, _Float16* __restrict__ hs)
{
  int b = blockIdx.x, c = blockIdx.y, n = threadIdx.x;
  float A = 1.0f / (1.0f + expf(-logA[n]));
  float P = prefix[(b * NCH + c) * N_STATE + n];
  long base = ((long)b * SS + (long)c * CHUNK) * N_STATE + n;
  float p = 1.f;
  for (int t = 0; t < CHUNK; ++t) {
    p *= A;
    hs[base + (long)t * N_STATE] = (_Float16)(bx[base + (long)t * N_STATE] + p * P);
  }
}

// ---------------- classifier head (tiny) ----------------
__global__ __launch_bounds__(256) void classifier_kernel(
    const float* __restrict__ hmean,
    const float* __restrict__ w1, const float* __restrict__ b1,
    const float* __restrict__ w2, const float* __restrict__ b2,
    float* __restrict__ out)
{
  __shared__ float sh[BB * D_MODEL];
  __shared__ float sh1[BB * D_MODEL];
  int t = threadIdx.x;
  for (int i = t; i < BB * D_MODEL; i += 256) sh[i] = hmean[i];
  __syncthreads();
  int e = t;
  for (int b = 0; b < BB; ++b) {
    float acc = b1[e];
    for (int d = 0; d < D_MODEL; ++d) acc += w1[e * D_MODEL + d] * sh[b * D_MODEL + d];
    sh1[b * D_MODEL + e] = acc / (1.0f + expf(-acc));   // silu
  }
  __syncthreads();
  if (t < BB * N_CLS) {
    int b = t / N_CLS, c = t % N_CLS;
    float acc = b2[c];
    for (int d = 0; d < D_MODEL; ++d) acc += w2[c * D_MODEL + d] * sh1[b * D_MODEL + d];
    out[t] = acc;
  }
}

// ---------------- host orchestration ----------------
extern "C" void kernel_launch(void* const* d_in, const int* in_sizes, int n_in,
                              void* d_out, int out_size, void* d_ws, size_t ws_size,
                              hipStream_t stream)
{
  const float* x      = (const float*)d_in[0];
  const float* in_w   = (const float*)d_in[1];
  const float* in_b   = (const float*)d_in[2];
  const float* log_A  = (const float*)d_in[3];
  const float* Bw     = (const float*)d_in[4];
  const float* Bb     = (const float*)d_in[5];
  const float* Cw     = (const float*)d_in[6];
  const float* Cb     = (const float*)d_in[7];
  const float* nscale = (const float*)d_in[8];
  const float* fscale = (const float*)d_in[9];
  const float* w1     = (const float*)d_in[10];
  const float* b1     = (const float*)d_in[11];
  const float* w2     = (const float*)d_in[12];
  const float* b2     = (const float*)d_in[13];
  float* out = (float*)d_out;

  char* ws = (char*)d_ws;
  size_t off = 0;
  auto alloc = [&](size_t bytes) -> void* {
    void* p = ws + off;
    off = (off + bytes + 255) & ~(size_t)255;
    return p;
  };
  float*    h    = (float*)   alloc((size_t)ROWS * D_MODEL * 4);   // 64 MiB
  _Float16* hn16 = (_Float16*)alloc((size_t)ROWS * D_MODEL * 2);   // 32 MiB
  _Float16* x16  = (_Float16*)hn16;                                // union: x16 only live before layer 0
  float*    bx   = (float*)   alloc((size_t)ROWS * N_STATE * 4);   // 8 MiB
  _Float16* hs16 = (_Float16*)alloc((size_t)ROWS * N_STATE * 2);   // 4 MiB
  _Float16* w1t  = (_Float16*)alloc((size_t)D_MODEL * D_IN * 2);
  _Float16* wbt  = (_Float16*)alloc((size_t)N_LAYERS * N_STATE * D_MODEL * 2);
  _Float16* wct  = (_Float16*)alloc((size_t)N_LAYERS * D_MODEL * N_STATE * 2);
  float* carry   = (float*)alloc((size_t)BB * NCH * N_STATE * 4);
  float* prefix  = (float*)alloc((size_t)BB * NCH * N_STATE * 4);
  float* rinv    = (float*)alloc((size_t)ROWS * 4);
  float* hmean   = (float*)alloc((size_t)BB * D_MODEL * 4);

  // Weight/activation f16 conversion. BT layouts [N][K] coincide with native
  // layouts of in_w [256][64], Bw [32][256], Cw [256][32] -> plain copies.
  auto cvt = [&](const float* s, _Float16* d, int n) {
    cvt_f16_kernel<<<(n + 255) / 256, 256, 0, stream>>>(s, d, n);
  };
  cvt(x,    x16, ROWS * D_IN);
  cvt(in_w, w1t, D_MODEL * D_IN);
  cvt(Bw,   wbt, N_LAYERS * N_STATE * D_MODEL);
  cvt(Cw,   wct, N_LAYERS * D_MODEL * N_STATE);

  // Input projection: h = x @ in_w^T + in_b
  wmma_gemm_kernel<<<dim3(ROWS / 128, D_MODEL / 32), 128, 0, stream>>>(
      x16, D_IN, w1t, D_IN, in_b, nullptr, 0, h, D_MODEL, D_IN);

  for (int l = 0; l < N_LAYERS; ++l) {
    rmsnorm_f16_kernel<<<ROWS, 256, 0, stream>>>(h, nscale + l * D_MODEL, hn16);

    // bx = hn @ Bw[l]^T + Bb[l]
    wmma_gemm_kernel<<<dim3(ROWS / 128, N_STATE / 32), 128, 0, stream>>>(
        hn16, D_MODEL, wbt + (size_t)l * N_STATE * D_MODEL, D_MODEL,
        Bb + l * N_STATE, nullptr, 0, bx, N_STATE, D_MODEL);

    // chunked diagonal scan
    scan_local_kernel<<<dim3(BB, NCH), N_STATE, 0, stream>>>(bx, log_A + l * N_STATE, carry);
    scan_combine_kernel<<<BB, N_STATE, 0, stream>>>(carry, log_A + l * N_STATE, prefix);
    scan_apply_kernel<<<dim3(BB, NCH), N_STATE, 0, stream>>>(bx, log_A + l * N_STATE, prefix, hs16);

    // h = hs @ Cw[l]^T + Cb[l] + residual(h)
    wmma_gemm_kernel<<<dim3(ROWS / 128, D_MODEL / 32), 128, 0, stream>>>(
        hs16, N_STATE, wct + (size_t)l * D_MODEL * N_STATE, N_STATE,
        Cb + l * D_MODEL, h, D_MODEL, h, D_MODEL, N_STATE);
  }

  rownorm_kernel<<<ROWS, 256, 0, stream>>>(h, rinv);
  mean_kernel<<<BB, 256, 0, stream>>>(h, rinv, fscale, hmean);
  classifier_kernel<<<1, 256, 0, stream>>>(hmean, w1, b1, w2, b2, out);
}
